// VectorQuantizer_44581760533186
// MI455X (gfx1250) — compile-verified, matrix operand order checked by reading, not run on hardware
//
#include <hip/hip_runtime.h>
#include <hip/hip_bf16.h>
#include <math.h>

typedef __attribute__((ext_vector_type(16))) _Float16 v16h;
typedef __attribute__((ext_vector_type(2)))  _Float16 v2h;
typedef __attribute__((ext_vector_type(8)))  float    v8f;
typedef __attribute__((ext_vector_type(4)))  unsigned int v4u;
typedef __attribute__((ext_vector_type(8)))  int      v8i;
typedef __attribute__((ext_vector_type(4)))  int      v4i;

// Problem constants (match reference setup_inputs)
#define BB 16
#define CC 256
#define HH 32
#define WW 32
#define NN (BB*HH*WW)   // 16384 tokens
#define KK 8192         // codebook entries
#define NTILES (NN/16)  // 1024
#define KTILES (KK/16)  // 512
#define KSPLIT 8        // codebook column slices (occupancy)
#define STAGE  4        // codebook tiles per TDM stage (32 KB)
#define NSTAGES ((KTILES/KSPLIT)/STAGE)   // 16
#define EPSN 1e-12f

// ---------------- norms ----------------

__global__ void z_norms_kernel(const float* __restrict__ z, float* __restrict__ inv_z) {
    int n = blockIdx.x * blockDim.x + threadIdx.x;          // 0..NN-1
    int b = n >> 10, hw = n & 1023;
    const float* p = z + (size_t)b * (CC*HH*WW) + hw;       // stride 1024 per channel
    float s = 0.f;
#pragma unroll 8
    for (int c = 0; c < CC; ++c) { float v = p[(size_t)c * 1024]; s += v * v; }
    inv_z[n] = 1.f / fmaxf(sqrtf(s), EPSN);
}

__global__ void e_norms_kernel(const float* __restrict__ e, float* __restrict__ inv_e) {
    int t = blockIdx.x * blockDim.x + threadIdx.x;
    int row = t >> 5, lane = t & 31;
    const float* p = e + (size_t)row * CC;
    float s = 0.f;
#pragma unroll
    for (int i = lane; i < CC; i += 32) { float v = p[i]; s += v * v; }
#pragma unroll
    for (int off = 16; off; off >>= 1) s += __shfl_xor(s, off, 32);
    if (lane == 0) inv_e[row] = 1.f / fmaxf(sqrtf(s), EPSN);
}

// ---------------- pack to WMMA operand order (f16) ----------------
// per (tile, kb): 32 lanes x 16 f16 contiguous per lane; dword index
// u = tile*2048 + kb*256 + lane*8 + v ;  k0(v,h)=kb*32+((v&4)?16:0)+h*8+(v&3)*2

__global__ void pack_z_kernel(const float* __restrict__ z, const float* __restrict__ invn,
                              v2h* __restrict__ out) {
    int u = blockIdx.x * blockDim.x + threadIdx.x;
    int tile = u >> 11, kb = (u >> 8) & 7, lane = (u >> 3) & 31, v = u & 7;
    int r = lane & 15, h = lane >> 4;
    int n = tile * 16 + r;
    int k0 = kb * 32 + ((v & 4) ? 16 : 0) + h * 8 + (v & 3) * 2;
    int b = n >> 10, hw = n & 1023;
    const float* p = z + (size_t)b * (CC*HH*WW) + hw;
    float s = invn[n];
    v2h o;
    o[0] = (_Float16)(p[(size_t)k0       * 1024] * s);
    o[1] = (_Float16)(p[(size_t)(k0 + 1) * 1024] * s);
    out[u] = o;
}

__global__ void pack_e_kernel(const float* __restrict__ e, const float* __restrict__ invn,
                              v2h* __restrict__ out) {
    int u = blockIdx.x * blockDim.x + threadIdx.x;
    int tile = u >> 11, kb = (u >> 8) & 7, lane = (u >> 3) & 31, v = u & 7;
    int r = lane & 15, h = lane >> 4;
    int n = tile * 16 + r;
    int k0 = kb * 32 + ((v & 4) ? 16 : 0) + h * 8 + (v & 3) * 2;
    const float* p = e + (size_t)n * CC;
    float s = invn[n];
    v2h o;
    o[0] = (_Float16)(p[k0]     * s);
    o[1] = (_Float16)(p[k0 + 1] * s);
    out[u] = o;
}

// ---------------- TDM: 1-D tile copy global -> LDS ----------------
__device__ static inline void tdm_load_1d(unsigned lds_addr, const void* gptr, unsigned nelem8) {
    unsigned long long ga = (unsigned long long)(size_t)gptr;
    v4u g0 = { 1u,                                   // count=1, no gather
               lds_addr,
               (unsigned)(ga & 0xFFFFFFFFu),
               (unsigned)(((ga >> 32) & 0x1FFFFFFu) | (2u << 30)) };  // type=2
    v8i g1 = { (int)(3u << 16),                      // data_size=8B, wg_mask=0
               (int)((nelem8 & 0xFFFFu) << 16),      // tensor_dim0 lo16
               (int)((nelem8 >> 16) | (1u << 16)),   // tensor_dim0 hi16 | tensor_dim1 lo16=1
               (int)((nelem8 & 0xFFFFu) << 16),      // tensor_dim1 hi=0 | tile_dim0
               (int)1,                               // tile_dim1=1, tile_dim2=0
               (int)nelem8,                          // tensor_dim0_stride lo32
               0, 0 };
    v4i z4 = {0, 0, 0, 0};
#if defined(__clang_major__) && (__clang_major__ >= 23)
    v8i z8 = {0, 0, 0, 0, 0, 0, 0, 0};
    __builtin_amdgcn_tensor_load_to_lds(g0, g1, z4, z4, z8, 0);
#else
    __builtin_amdgcn_tensor_load_to_lds(g0, g1, z4, z4, 0);
#endif
}

// ---------------- main GEMM + fused argmax ----------------
// block = 8 waves; each wave owns 2 M-tiles; block shares TDM-staged B (2x32KB).
// grid: 512 blocks = 64 M-groups x 8 K-slices. Partial (val,idx) per slice.
__global__ void __launch_bounds__(256)
vq_argmax_kernel(const v16h* __restrict__ a_packed, const v16h* __restrict__ e_packed,
                 float* __restrict__ pval, int* __restrict__ pidx) {
    __shared__ v16h smem[2 * STAGE * 8 * 32];               // 2 x 32 KB

    int mg   = blockIdx.x & 63;                             // M-group
    int ks   = blockIdx.x >> 6;                             // K-slice
    int wib  = threadIdx.x >> 5;                            // wave in block
    int lane = threadIdx.x & 31;
    int m0   = (mg * 8 + wib) * 2;                          // two token tiles
    int m1   = m0 + 1;
    int ktile0 = ks * (KTILES / KSPLIT);                    // first codebook tile of slice

    // A in registers: 2 tiles x 8 kb-blocks = 128 VGPRs
    v16h a0[8], a1[8];
    const v16h* ap0 = a_packed + (size_t)m0 * 256 + lane;
    const v16h* ap1 = a_packed + (size_t)m1 * 256 + lane;
#pragma unroll
    for (int kb = 0; kb < 8; ++kb) { a0[kb] = ap0[kb * 32]; a1[kb] = ap1[kb * 32]; }

    float best0[8], best1[8];
    int   bidx0[8], bidx1[8];
#pragma unroll
    for (int r = 0; r < 8; ++r) {
        best0[r] = -3.402823466e+38f; bidx0[r] = 0;
        best1[r] = -3.402823466e+38f; bidx1[r] = 0;
    }

    // prologue: stage 0 -> buffer 0
    if (wib == 0) {
        tdm_load_1d((unsigned)(size_t)&smem[0],
                    e_packed + (size_t)(ktile0) * 256, STAGE * 8 * 32 * 4);
    }

    for (int s = 0; s < NSTAGES; ++s) {
        int cur = s & 1;
        __syncthreads();                                    // all done reading buf[1^cur]
        if (wib == 0) {
            if (s + 1 < NSTAGES) {
                tdm_load_1d((unsigned)(size_t)&smem[(1 ^ cur) * STAGE * 8 * 32],
                            e_packed + (size_t)(ktile0 + (s + 1) * STAGE) * 256,
                            STAGE * 8 * 32 * 4);
                __builtin_amdgcn_s_wait_tensorcnt(1);       // stage s arrived
            } else {
                __builtin_amdgcn_s_wait_tensorcnt(0);
            }
        }
        __syncthreads();                                    // buf[cur] ready for all

        // software-pipelined B: load b(kb+1) before the WMMAs on b(kb)
        const v16h* bpl = &smem[cur * STAGE * 8 * 32] + lane;
        v16h bcur = bpl[0];
#pragma unroll
        for (int tt = 0; tt < STAGE; ++tt) {
            v8f acc0 = {}, acc1 = {};
#pragma unroll
            for (int kb = 0; kb < 8; ++kb) {
                const int nxt = tt * 8 + kb + 1;
                v16h bnext = bcur;
                if (nxt < STAGE * 8) bnext = bpl[nxt * 32];
                acc0 = __builtin_amdgcn_wmma_f32_16x16x32_f16(
                    false, a0[kb], false, bcur, (short)0, acc0, false, false);
                acc1 = __builtin_amdgcn_wmma_f32_16x16x32_f16(
                    false, a1[kb], false, bcur, (short)0, acc1, false, false);
                bcur = bnext;
            }
            int col = (ktile0 + s * STAGE + tt) * 16 + (lane & 15);
#pragma unroll
            for (int r = 0; r < 8; ++r) {
                if (acc0[r] > best0[r]) { best0[r] = acc0[r]; bidx0[r] = col; }
                if (acc1[r] > best1[r]) { best1[r] = acc1[r]; bidx1[r] = col; }
            }
        }
    }

    // cross-lane argmax within 16-lane halves (ties -> smaller idx)
#pragma unroll
    for (int r = 0; r < 8; ++r) {
        float v0 = best0[r]; int i0 = bidx0[r];
        float v1 = best1[r]; int i1 = bidx1[r];
#pragma unroll
        for (int off = 1; off < 16; off <<= 1) {
            float ov0 = __shfl_xor(v0, off, 32); int oi0 = __shfl_xor(i0, off, 32);
            float ov1 = __shfl_xor(v1, off, 32); int oi1 = __shfl_xor(i1, off, 32);
            if (ov0 > v0 || (ov0 == v0 && oi0 < i0)) { v0 = ov0; i0 = oi0; }
            if (ov1 > v1 || (ov1 == v1 && oi1 < i1)) { v1 = ov1; i1 = oi1; }
        }
        if ((lane & 15) == r) {
            int row = r + 8 * (lane >> 4);
            int n0 = m0 * 16 + row, n1 = m1 * 16 + row;
            pval[(size_t)ks * NN + n0] = v0; pidx[(size_t)ks * NN + n0] = i0;
            pval[(size_t)ks * NN + n1] = v1; pidx[(size_t)ks * NN + n1] = i1;
        }
    }
}

// merge K-slice partials; ascending ks + strict > keeps first-min tie rule
__global__ void argmax_reduce_kernel(const float* __restrict__ pval,
                                     const int* __restrict__ pidx,
                                     int* __restrict__ idx) {
    int n = blockIdx.x * blockDim.x + threadIdx.x;          // 0..NN-1
    float bv = pval[n];
    int   bi = pidx[n];
#pragma unroll
    for (int ks = 1; ks < KSPLIT; ++ks) {
        float v = pval[(size_t)ks * NN + n];
        int   i = pidx[(size_t)ks * NN + n];
        if (v > bv) { bv = v; bi = i; }
    }
    idx[n] = bi;
}

// ---------------- epilogue: gather + normalize + transpose, emit indices ----------------
__global__ void vq_output_kernel(const float* __restrict__ emb, const float* __restrict__ inv_e,
                                 const int* __restrict__ idx, float* __restrict__ out) {
    int t = blockIdx.x * blockDim.x + threadIdx.x;          // == out offset
    int hw = t & 1023;
    int c  = (t >> 10) & 255;
    int b  = t >> 18;
    int n  = b * 1024 + hw;
    int k  = idx[n];
    out[t] = emb[(size_t)k * CC + c] * inv_e[k];
    if (t < NN) {
        out[(size_t)NN * CC + t] = (float)idx[t];           // indices [B,H,W]
    }
}

extern "C" void kernel_launch(void* const* d_in, const int* in_sizes, int n_in,
                              void* d_out, int out_size, void* d_ws, size_t ws_size,
                              hipStream_t stream) {
    const float* z   = (const float*)d_in[0];   // [16,256,32,32]
    const float* emb = (const float*)d_in[1];   // [8192,256]
    float* out = (float*)d_out;

    char* ws = (char*)d_ws;
    v2h*   a_packed = (v2h*)(ws);                               //  8 MB
    v2h*   e_packed = (v2h*)(ws + 8388608);                     //  4 MB
    float* inv_e    = (float*)(ws + 12582912);                  // 32 KB
    float* inv_z    = (float*)(ws + 12615680);                  // 64 KB
    int*   idx      = (int*)(ws + 12681216);                    // 64 KB
    float* pval     = (float*)(ws + 12746752);                  // 512 KB
    int*   pidx     = (int*)(ws + 13271040);                    // 512 KB

    z_norms_kernel<<<NN / 256, 256, 0, stream>>>(z, inv_z);
    e_norms_kernel<<<(KK * 32) / 256, 256, 0, stream>>>(emb, inv_e);
    pack_z_kernel<<<(NN * CC / 2) / 256, 256, 0, stream>>>(z, inv_z, a_packed);
    pack_e_kernel<<<(KK * CC / 2) / 256, 256, 0, stream>>>(emb, inv_e, e_packed);
    vq_argmax_kernel<<<64 * KSPLIT, 256, 0, stream>>>(
        (const v16h*)a_packed, (const v16h*)e_packed, pval, pidx);
    argmax_reduce_kernel<<<NN / 256, 256, 0, stream>>>(pval, pidx, idx);
    vq_output_kernel<<<(NN * CC) / 256, 256, 0, stream>>>(emb, inv_e, idx, out);
}
